// LSTMDecoderWithBahdanauAttention_2156073583150
// MI455X (gfx1250) — matrix operand order repaired and checked by reading, not compiled, not run on hardware
//
#include <hip/hip_runtime.h>

// ---------------------------------------------------------------------------
// LSTM decoder with Bahdanau attention for MI455X (gfx1250, wave32, WMMA).
// bf16 inputs + f32 accumulation via v_wmma_f32_16x16x32_bf16.
// Gates GEMM stages activations to LDS via global_load_async_to_lds_b128.
// ---------------------------------------------------------------------------

typedef __bf16 bf16;
typedef __attribute__((ext_vector_type(16))) __bf16 v16bf;
typedef __attribute__((ext_vector_type(8)))  __bf16 v8bf;
typedef __attribute__((ext_vector_type(8)))  float  v8f;

// exact pointee type expected by __builtin_amdgcn_global_load_async_to_lds_b128
typedef int v4i __attribute__((vector_size(16)));
typedef __attribute__((address_space(1))) v4i glob_v4i;
typedef __attribute__((address_space(3))) v4i lds_v4i;

#define B_SZ    64
#define T_SZ    128
#define IN_DIM  64
#define T_ENC   256
#define HID     512
#define ATTN    256
#define OUT_DIM 64
#define G4H     (4 * HID)          // 2048 gate columns (i,f,g,o)
#define INPW    (IN_DIM + HID)     // 576 = concat(x_t, context)

union FragBF { v16bf v; v8bf h[2]; };

// A fragment, 16x32 bf16 (MxK), row-major A with leading dim lda.
// Layout: lanes 0-15 (g=0): K = k0+0..15;
//         lanes 16-31 (g=1): K = k0+16..31.  Two 16B loads.
__device__ __forceinline__ v16bf load_a_frag(const bf16* __restrict__ A, int lda,
                                             int m0, int k0, int lane) {
  int m = lane & 15, g = lane >> 4;
  const bf16* p = A + (size_t)(m0 + m) * lda + k0 + g * 16;
  FragBF u;
  u.h[0] = *(const v8bf*)(p);
  u.h[1] = *(const v8bf*)(p + 8);
  return u.v;
}

// B fragment, 32x16 bf16 (KxN) with B = W^T, W row-major (N rows, K cols).
// Lane group g holds K = k0+g*16 .. +15 for column n0+(lane&15): one
// contiguous 32B run of W row (n0+n).  Two 16B loads.
__device__ __forceinline__ v16bf load_b_frag_wt(const bf16* __restrict__ W, int ldw,
                                                int k0, int n0, int lane) {
  int n = lane & 15, g = lane >> 4;
  const bf16* p = W + (size_t)(n0 + n) * ldw + k0 + g * 16;
  FragBF u;
  u.h[0] = *(const v8bf*)(p);
  u.h[1] = *(const v8bf*)(p + 8);
  return u.v;
}

// C/D f32 16x16: VGPR r = row (r + 8*g), column = lane&15.
__device__ __forceinline__ void store_c_frag(float* __restrict__ C, int ldc,
                                             int m0, int n0, int lane, v8f acc,
                                             const float* __restrict__ bias) {
  int n = lane & 15, g = lane >> 4;
  float bv = bias ? bias[n0 + n] : 0.0f;
#pragma unroll
  for (int r = 0; r < 8; ++r)
    C[(size_t)(m0 + r + g * 8) * ldc + n0 + n] = acc[r] + bv;
}

__device__ __forceinline__ v8f wmma_bf16(v16bf a, v16bf b, v8f c) {
  return __builtin_amdgcn_wmma_f32_16x16x32_bf16(false, a, false, b, (short)0, c,
                                                 false, false);
}

// ---------------------------------------------------------------------------
// Generic C[M,N] = A[M,K](bf16) * W[N,K]^T + bias, one wave = 16x64 strip
// (4 N-tiles reusing one A fragment => 2.5 loads per WMMA instead of 4).
// Requires N % 64 == 0; grid sized to exact strip count (EXEC all-ones).
// ---------------------------------------------------------------------------
__global__ __launch_bounds__(256)
void gemm_bf16nt_x4_kernel(const bf16* __restrict__ A, const bf16* __restrict__ W,
                           float* __restrict__ C, const float* __restrict__ bias,
                           int M, int N, int K, int lda, int ldw, int ldc) {
  int wave = (int)((blockIdx.x * blockDim.x + threadIdx.x) >> 5);
  int lane = (int)(threadIdx.x & 31);
  int ng_count = N >> 6;                     // 64-wide N strips
  int tm = wave / ng_count, ng = wave - tm * ng_count;
  int m0 = tm << 4, n0 = ng << 6;
  if (m0 >= M) return;                        // wave-uniform guard
  v8f acc0 = {}, acc1 = {}, acc2 = {}, acc3 = {};
  for (int k = 0; k < K; k += 32) {
    v16bf a = load_a_frag(A, lda, m0, k, lane);
    __builtin_prefetch(W + (size_t)(n0 + (lane & 15)) * ldw + k + 64, 0, 0);
    acc0 = wmma_bf16(a, load_b_frag_wt(W, ldw, k, n0,      lane), acc0);
    acc1 = wmma_bf16(a, load_b_frag_wt(W, ldw, k, n0 + 16, lane), acc1);
    acc2 = wmma_bf16(a, load_b_frag_wt(W, ldw, k, n0 + 32, lane), acc2);
    acc3 = wmma_bf16(a, load_b_frag_wt(W, ldw, k, n0 + 48, lane), acc3);
  }
  store_c_frag(C, ldc, m0, n0,      lane, acc0, bias);
  store_c_frag(C, ldc, m0, n0 + 16, lane, acc1, bias);
  store_c_frag(C, ldc, m0, n0 + 32, lane, acc2, bias);
  store_c_frag(C, ldc, m0, n0 + 48, lane, acc3, bias);
}

// ---------------------------------------------------------------------------
// gates[64,2048] = inp[64,576]*Wih^T + h[64,512]*Whh^T + (b_ih+b_hh).
// 16 blocks x 8 waves; all waves in a block share one M-tile, whose 16 rows of
// inp and h are staged into LDS with async global->LDS (ASYNCcnt path).
// Each wave owns a 16x64 strip: 34 k-steps x 4 WMMAs, A from LDS, B streamed.
// ---------------------------------------------------------------------------
__global__ __launch_bounds__(256)
void gates_kernel(const bf16* __restrict__ inp, const bf16* __restrict__ Wih,
                  const bf16* __restrict__ hbf, const bf16* __restrict__ Whh,
                  const float* __restrict__ bias, float* __restrict__ gates) {
  __shared__ alignas(16) bf16 sA[16 * INPW + 16 * HID];   // 34.8 KB
  bf16* sInp = sA;
  bf16* sH   = sA + 16 * INPW;

  int tid = (int)threadIdx.x;
  int tm  = (int)blockIdx.x >> 2;            // 0..3  (M tile)
  int ng0 = ((int)blockIdx.x & 3) * 8;       // base 64-wide N strip
  int m0  = tm << 4;

  const bf16* gInp = inp + (size_t)m0 * INPW;   // 16 contiguous rows
  const bf16* gH   = hbf + (size_t)m0 * HID;

  const int INP_CH = (16 * INPW) / 8;        // 16B chunks: 1152
  const int HID_CH = (16 * HID) / 8;         // 1024

#if __has_builtin(__builtin_amdgcn_global_load_async_to_lds_b128)
  for (int i = tid; i < INP_CH; i += 256)
    __builtin_amdgcn_global_load_async_to_lds_b128(
        (glob_v4i*)(gInp + i * 8), (lds_v4i*)(sInp + i * 8), 0, 0);
  for (int i = tid; i < HID_CH; i += 256)
    __builtin_amdgcn_global_load_async_to_lds_b128(
        (glob_v4i*)(gH + i * 8), (lds_v4i*)(sH + i * 8), 0, 0);
#if __has_builtin(__builtin_amdgcn_s_wait_asynccnt)
  __builtin_amdgcn_s_wait_asynccnt(0);
#else
  asm volatile("s_wait_asynccnt 0x0" ::: "memory");
#endif
#else
  for (int i = tid; i < INP_CH; i += 256)
    *(v8bf*)(sInp + i * 8) = *(const v8bf*)(gInp + i * 8);
  for (int i = tid; i < HID_CH; i += 256)
    *(v8bf*)(sH + i * 8) = *(const v8bf*)(gH + i * 8);
#endif
  __syncthreads();

  int wave = tid >> 5, lane = tid & 31;
  int n0 = (ng0 + wave) << 6;                // this wave's 64-col strip

  v8f acc0 = {}, acc1 = {}, acc2 = {}, acc3 = {};
  for (int k = 0; k < INPW; k += 32) {       // 18 k-steps, A from LDS
    v16bf a = load_a_frag(sInp, INPW, 0, k, lane);
    __builtin_prefetch(Wih + (size_t)(n0 + (lane & 15)) * INPW + k + 64, 0, 0);
    acc0 = wmma_bf16(a, load_b_frag_wt(Wih, INPW, k, n0,      lane), acc0);
    acc1 = wmma_bf16(a, load_b_frag_wt(Wih, INPW, k, n0 + 16, lane), acc1);
    acc2 = wmma_bf16(a, load_b_frag_wt(Wih, INPW, k, n0 + 32, lane), acc2);
    acc3 = wmma_bf16(a, load_b_frag_wt(Wih, INPW, k, n0 + 48, lane), acc3);
  }
  for (int k = 0; k < HID; k += 32) {        // 16 k-steps, A from LDS
    v16bf a = load_a_frag(sH, HID, 0, k, lane);
    __builtin_prefetch(Whh + (size_t)(n0 + (lane & 15)) * HID + k + 64, 0, 0);
    acc0 = wmma_bf16(a, load_b_frag_wt(Whh, HID, k, n0,      lane), acc0);
    acc1 = wmma_bf16(a, load_b_frag_wt(Whh, HID, k, n0 + 16, lane), acc1);
    acc2 = wmma_bf16(a, load_b_frag_wt(Whh, HID, k, n0 + 32, lane), acc2);
    acc3 = wmma_bf16(a, load_b_frag_wt(Whh, HID, k, n0 + 48, lane), acc3);
  }
  store_c_frag(gates, G4H, m0, n0,      lane, acc0, bias);
  store_c_frag(gates, G4H, m0, n0 + 16, lane, acc1, bias);
  store_c_frag(gates, G4H, m0, n0 + 32, lane, acc2, bias);
  store_c_frag(gates, G4H, m0, n0 + 48, lane, acc3, bias);
}

// ---------------------------------------------------------------------------
// Bahdanau attention: one block per batch element (256 threads).
// ---------------------------------------------------------------------------
__global__ __launch_bounds__(256)
void attn_kernel(const float* __restrict__ q, const float* __restrict__ keys,
                 const float* __restrict__ vvec, const bf16* __restrict__ enc,
                 const bf16* __restrict__ xbf, int t, bf16* __restrict__ inp) {
  int b = blockIdx.x;
  int tid = threadIdx.x;
  __shared__ float sq[ATTN];
  __shared__ float sv[ATTN];
  __shared__ float sred[T_ENC];
  __shared__ float sw[T_ENC];

  sq[tid] = q[(size_t)b * ATTN + tid];
  sv[tid] = vvec[tid];
  __syncthreads();

  const float* krow = keys + ((size_t)b * T_ENC + tid) * ATTN;
  float acc = 0.0f;
#pragma unroll 4
  for (int a = 0; a < ATTN; ++a)
    acc += tanhf(sq[a] + krow[a]) * sv[a];

  sred[tid] = acc;
  __syncthreads();
  for (int s = 128; s > 0; s >>= 1) {
    if (tid < s) sred[tid] = fmaxf(sred[tid], sred[tid + s]);
    __syncthreads();
  }
  float mx = sred[0];
  __syncthreads();
  float ex = __expf(acc - mx);
  sred[tid] = ex;
  __syncthreads();
  for (int s = 128; s > 0; s >>= 1) {
    if (tid < s) sred[tid] += sred[tid + s];
    __syncthreads();
  }
  float w = ex / sred[0];
  sw[tid] = w;
  __syncthreads();

  float c0 = 0.0f, c1 = 0.0f;
  for (int e = 0; e < T_ENC; ++e) {
    float we = sw[e];
    const bf16* er = enc + ((size_t)b * T_ENC + e) * HID;
    c0 += we * (float)er[tid];
    c1 += we * (float)er[tid + 256];
  }
  bf16* row = inp + (size_t)b * INPW;
  row[IN_DIM + tid]       = (bf16)c0;
  row[IN_DIM + tid + 256] = (bf16)c1;
  if (tid < IN_DIM)
    row[tid] = xbf[((size_t)b * T_SZ + t) * IN_DIM + tid];
}

// ---------------------------------------------------------------------------
// LSTM pointwise: c' = sig(f)*c + sig(i)*tanh(g); h' = sig(o)*tanh(c').
// ---------------------------------------------------------------------------
__global__ __launch_bounds__(256)
void lstm_pointwise_kernel(const float* __restrict__ gates, float* __restrict__ c,
                           bf16* __restrict__ hbf) {
  int idx = (int)(blockIdx.x * blockDim.x + threadIdx.x);
  if (idx >= B_SZ * HID) return;
  int b = idx >> 9, j = idx & (HID - 1);
  const float* g = gates + (size_t)b * G4H;
  float ig = 1.0f / (1.0f + __expf(-g[j]));
  float fg = 1.0f / (1.0f + __expf(-g[j + HID]));
  float gg = tanhf(g[j + 2 * HID]);
  float og = 1.0f / (1.0f + __expf(-g[j + 3 * HID]));
  float cn = fg * c[idx] + ig * gg;
  c[idx] = cn;
  hbf[idx] = (bf16)(og * tanhf(cn));
}

// ---------------------------------------------------------------------------
// Helpers: f32 -> bf16 conversion, state/bias init.
// ---------------------------------------------------------------------------
__global__ __launch_bounds__(256)
void cvt_f32_to_bf16(const float* __restrict__ src, bf16* __restrict__ dst, int n) {
  int i = (int)(blockIdx.x * blockDim.x + threadIdx.x);
  if (i < n) dst[i] = (bf16)src[i];
}

__global__ __launch_bounds__(256)
void init_kernel(float* __restrict__ c, bf16* __restrict__ hbf,
                 float* __restrict__ bias, const float* __restrict__ b_ih,
                 const float* __restrict__ b_hh) {
  int i = (int)(blockIdx.x * blockDim.x + threadIdx.x);
  if (i < B_SZ * HID) { c[i] = 0.0f; hbf[i] = (bf16)0.0f; }
  if (i < G4H) bias[i] = b_ih[i] + b_hh[i];
}

// ---------------------------------------------------------------------------
extern "C" void kernel_launch(void* const* d_in, const int* in_sizes, int n_in,
                              void* d_out, int out_size, void* d_ws, size_t ws_size,
                              hipStream_t stream) {
  const float* x    = (const float*)d_in[0];   // (B, T, IN)
  const float* enc  = (const float*)d_in[1];   // (B, T_ENC, HID)
  const float* Wq   = (const float*)d_in[2];   // (ATTN, HID)
  const float* Wk   = (const float*)d_in[3];   // (ATTN, HID)
  const float* v    = (const float*)d_in[4];   // (1, ATTN)
  const float* Wih  = (const float*)d_in[5];   // (4H, INPW)
  const float* Whh  = (const float*)d_in[6];   // (4H, HID)
  const float* b_ih = (const float*)d_in[7];   // (4H,)
  const float* b_hh = (const float*)d_in[8];   // (4H,)
  const float* Wfc  = (const float*)d_in[9];   // (OUT, HID)
  const float* b_fc = (const float*)d_in[10];  // (OUT,)
  float* out = (float*)d_out;                  // (B, T, OUT)

  // ---- workspace carve-up (256B aligned) ----
  char* ws = (char*)d_ws;
  size_t off = 0;
  auto carve = [&](size_t bytes) -> void* {
    void* p = ws + off;
    off = (off + bytes + 255) & ~(size_t)255;
    return p;
  };
  bf16*  enc_bf = (bf16*) carve(sizeof(bf16) * (size_t)B_SZ * T_ENC * HID);
  bf16*  x_bf   = (bf16*) carve(sizeof(bf16) * (size_t)B_SZ * T_SZ * IN_DIM);
  bf16*  Wq_bf  = (bf16*) carve(sizeof(bf16) * (size_t)ATTN * HID);
  bf16*  Wk_bf  = (bf16*) carve(sizeof(bf16) * (size_t)ATTN * HID);
  bf16*  Wih_bf = (bf16*) carve(sizeof(bf16) * (size_t)G4H * INPW);
  bf16*  Whh_bf = (bf16*) carve(sizeof(bf16) * (size_t)G4H * HID);
  bf16*  Wfc_bf = (bf16*) carve(sizeof(bf16) * (size_t)OUT_DIM * HID);
  float* keys   = (float*)carve(sizeof(float) * (size_t)B_SZ * T_ENC * ATTN);
  float* qbuf   = (float*)carve(sizeof(float) * (size_t)B_SZ * ATTN);
  float* gates  = (float*)carve(sizeof(float) * (size_t)B_SZ * G4H);
  float* c_st   = (float*)carve(sizeof(float) * (size_t)B_SZ * HID);
  bf16*  h_bf   = (bf16*) carve(sizeof(bf16) * (size_t)B_SZ * HID);
  bf16*  inp_bf = (bf16*) carve(sizeof(bf16) * (size_t)B_SZ * INPW);
  float* bias   = (float*)carve(sizeof(float) * (size_t)G4H);

  auto cvt = [&](const float* s, bf16* d, int n) {
    cvt_f32_to_bf16<<<(n + 255) / 256, 256, 0, stream>>>(s, d, n);
  };

  // ---- one-time: precision conversion, bias fuse, state init, keys GEMM ----
  cvt(enc, enc_bf, B_SZ * T_ENC * HID);
  cvt(x,   x_bf,   B_SZ * T_SZ * IN_DIM);
  cvt(Wq,  Wq_bf,  ATTN * HID);
  cvt(Wk,  Wk_bf,  ATTN * HID);
  cvt(Wih, Wih_bf, G4H * INPW);
  cvt(Whh, Whh_bf, G4H * HID);
  cvt(Wfc, Wfc_bf, OUT_DIM * HID);
  init_kernel<<<(B_SZ * HID + 255) / 256, 256, 0, stream>>>(c_st, h_bf, bias, b_ih, b_hh);

  // keys = enc * Wk^T : M=16384, N=256 -> (1024 tm)*(4 strips) = 4096 waves
  gemm_bf16nt_x4_kernel<<<512, 256, 0, stream>>>(enc_bf, Wk_bf, keys, nullptr,
                                                 B_SZ * T_ENC, ATTN, HID,
                                                 HID, HID, ATTN);

  // ---- recurrence: T dependent steps, stream-ordered ----
  for (int t = 0; t < T_SZ; ++t) {
    // q = h * Wq^T : (4 tm)*(4 strips) = 16 waves = 2 blocks
    gemm_bf16nt_x4_kernel<<<2, 256, 0, stream>>>(h_bf, Wq_bf, qbuf, nullptr,
                                                 B_SZ, ATTN, HID, HID, HID, ATTN);
    // attention + build concat(x_t, context)
    attn_kernel<<<B_SZ, 256, 0, stream>>>(qbuf, keys, v, enc_bf, x_bf, t, inp_bf);
    // gates: 16 blocks x 8 waves, LDS-staged A
    gates_kernel<<<16, 256, 0, stream>>>(inp_bf, Wih_bf, h_bf, Whh_bf, bias, gates);
    // LSTM cell update (writes c and h_bf for next step)
    lstm_pointwise_kernel<<<(B_SZ * HID + 255) / 256, 256, 0, stream>>>(gates, c_st, h_bf);
    // out[:, t, :] = h * Wfc^T + b_fc : (4 tm)*(1 strip) = 4 waves = 1 block/128
    gemm_bf16nt_x4_kernel<<<1, 128, 0, stream>>>(h_bf, Wfc_bf, out + (size_t)t * OUT_DIM,
                                                 b_fc, B_SZ, OUT_DIM, HID,
                                                 HID, HID, T_SZ * OUT_DIM);
  }
}
